// SingleScaleGrid_45724221833359
// MI455X (gfx1250) — matrix operand order repaired and checked by reading, not compiled, not run on hardware
//
#include <hip/hip_runtime.h>
#include <hip/hip_bf16.h>
#include <math.h>

typedef __attribute__((ext_vector_type(16))) _Float16 v16h;
typedef __attribute__((ext_vector_type(2)))  _Float16 h2;
typedef __attribute__((ext_vector_type(8)))  float    v8f;

#define N_RAYSC    4096
#define N_SAMP     256
#define RESC       128
#define RES3       (RESC*RESC*RESC)
#define NEARV      0.2f
#define FARV       3.0f
#define STEPSZ     0.5f
#define BGV        1.0f
#define ACT_SHIFT  (-4.59511985013459f)
#define NFRAG      26

// output layout (floats): rgb[4096*3] | depth[4096] | disp[4096] | acc[4096]
//                         | normals[4096*3] | weights[4096*256] | alphainv_cum[4096*257]
#define OFF_DEPTH  12288
#define OFF_DISP   16384
#define OFF_ACC    20480
#define OFF_NRM    24576
#define OFF_W      36864
#define OFF_AIC    1085440

struct SMem {
  alignas(16) _Float16 wfrag[NFRAG * 512];     // 26624 B : WMMA B fragments
  alignas(16) _Float16 buf[N_SAMP * 64];       // 32768 B : activations (in-place per layer)
  float bias[3 * 64 + 16];
  float alphaS[N_SAMP];
  float nrmS[N_SAMP][3];
  float rgbS[N_SAMP][3];
  float waveSum[8];
  float red[8][8];
  float Ttot;
};

// A fragment (16x32 f16): lane M = lane&15, halfgroup h = lane>>4.
// halves j=0..7 -> K = kb*32 + h*8 + j ; j=8..15 -> K = kb*32 + 16 + h*8 + (j-8)
__device__ __forceinline__ v16h ld_fragA(const _Float16* rowp, int kb, int h) {
  union { v16h v; float4 f[2]; } u;
  u.f[0] = *(const float4*)(rowp + kb * 32 + h * 8);
  u.f[1] = *(const float4*)(rowp + kb * 32 + 16 + h * 8);
  return u.v;
}

// B fragment: each lane's 16 halves pre-swizzled contiguously (32 B)
__device__ __forceinline__ v16h ld_fragB(const _Float16* p) {
  union { v16h v; float4 f[2]; } u;
  u.f[0] = *(const float4*)(p);
  u.f[1] = *(const float4*)(p + 8);
  return u.v;
}

__global__ void __launch_bounds__(256)
nerf_render_fused(const float* __restrict__ rays_o, const float* __restrict__ rays_d,
                  const float* __restrict__ viewdirs, const float* __restrict__ latent,
                  const float* __restrict__ normals,
                  const float* __restrict__ W0, const float* __restrict__ b0,
                  const float* __restrict__ W1, const float* __restrict__ b1,
                  const float* __restrict__ W2, const float* __restrict__ b2,
                  const float* __restrict__ W3, const float* __restrict__ b3,
                  float* __restrict__ out)
{
  __shared__ SMem sm;
  const int tid  = threadIdx.x;
  const int lane = tid & 31;
  const int wv   = tid >> 5;
  const int ray  = blockIdx.x;

  // ---------------- stage MLP weights as WMMA B fragments in LDS ----------------
  {
    const float* Ws[4] = {W0, W1, W2, W3};
    for (int tsk = tid; tsk < NFRAG * 32; tsk += 256) {
      int frag = tsk >> 5, ln = tsk & 31;
      int layer, ntile, kb;
      if (frag < 24) { layer = frag >> 3; ntile = (frag & 7) >> 1; kb = frag & 1; }
      else           { layer = 3; ntile = 0; kb = frag - 24; }
      const float* W = Ws[layer];
      int Krows = (layer == 0) ? 54 : 64;
      int Ncols = (layer == 3) ? 3 : 64;
      int ld    = (layer == 3) ? 3 : 64;
      int n = ntile * 16 + (ln & 15);
      int h = ln >> 4;                       // K half-block selector
      _Float16* dst = sm.wfrag + frag * 512 + ln * 16;
#pragma unroll
      for (int j = 0; j < 16; ++j) {
        int k = kb * 32 + ((j < 8) ? (h * 8 + j) : (16 + h * 8 + (j - 8)));
        float v = (k < Krows && n < Ncols) ? W[k * ld + n] : 0.f;
        dst[j] = (_Float16)v;
      }
    }
    if (tid < 64) { sm.bias[tid] = b0[tid]; sm.bias[64 + tid] = b1[tid]; sm.bias[128 + tid] = b2[tid]; }
    if (tid < 16) sm.bias[192 + tid] = (tid < 3) ? b3[tid] : 0.f;
  }

  // ---------------- per-sample features (1 thread = 1 sample) ----------------
  float t_s, alpha_s, nxs, nys, nzs;
  {
    const int s = tid;
    float ox = rays_o[ray*3+0], oy = rays_o[ray*3+1], oz = rays_o[ray*3+2];
    float dx = rays_d[ray*3+0], dy = rays_d[ray*3+1], dz = rays_d[ray*3+2];
    float di = rsqrtf(dx*dx + dy*dy + dz*dz); dx *= di; dy *= di; dz *= di;
    float vx = viewdirs[ray*3+0], vy = viewdirs[ray*3+1], vz = viewdirs[ray*3+2];
    float vi = rsqrtf(vx*vx + vy*vy + vz*vz); vx *= vi; vy *= vi; vz *= vi;

    float t = NEARV + (FARV - NEARV) * ((float)s * (1.0f / 255.0f));
    t_s = t;
    float px = ox + dx * t, py = oy + dy * t, pz = oz + dz * t;
    bool inb = (px >= -1.f && px <= 1.f && py >= -1.f && py <= 1.f && pz >= -1.f && pz <= 1.f);

    float qx = fminf(fmaxf((px + 1.f) * 0.5f, 0.f), 1.f) * 127.f;
    float qy = fminf(fmaxf((py + 1.f) * 0.5f, 0.f), 1.f) * 127.f;
    float qz = fminf(fmaxf((pz + 1.f) * 0.5f, 0.f), 1.f) * 127.f;
    int ix = min((int)floorf(qx), 126);
    int iy = min((int)floorf(qy), 126);
    int iz = min((int)floorf(qz), 126);
    float fx = qx - (float)ix, fy = qy - (float)iy, fz = qz - (float)iz;
    float gx = 1.f - fx, gy = 1.f - fy, gz = 1.f - fz;
    int base = (ix * RESC + iy) * RESC + iz;
    float w000 = gx*gy*gz, w001 = gx*gy*fz, w010 = gx*fy*gz, w011 = gx*fy*fz;
    float w100 = fx*gy*gz, w101 = fx*gy*fz, w110 = fx*fy*gz, w111 = fx*fy*fz;

    float lat[16];
#pragma unroll
    for (int c = 0; c < 16; ++c) {
      const float* p = latent + c * RES3 + base;
      lat[c] = w000*p[0] + w001*p[1] + w010*p[RESC] + w011*p[RESC+1]
             + w100*p[RESC*RESC] + w101*p[RESC*RESC+1]
             + w110*p[RESC*RESC+RESC] + w111*p[RESC*RESC+RESC+1];
    }
    float nrm[3];
#pragma unroll
    for (int c = 0; c < 3; ++c) {
      const float* p = normals + c * RES3 + base;
      nrm[c] = w000*p[0] + w001*p[1] + w010*p[RESC] + w011*p[RESC+1]
             + w100*p[RESC*RESC] + w101*p[RESC*RESC+1]
             + w110*p[RESC*RESC+RESC] + w111*p[RESC*RESC+RESC+1];
    }
    float nl = sqrtf(nrm[0]*nrm[0] + nrm[1]*nrm[1] + nrm[2]*nrm[2]);
    float ni = 1.f / fmaxf(nl, 1e-12f);
    float n0 = nrm[0]*ni, n1 = nrm[1]*ni, n2 = nrm[2]*ni;
    if (!inb) { n0 = 0.f; n1 = 0.f; n2 = 0.f; }
    nxs = n0; nys = n1; nzs = n2;
    sm.nrmS[s][0] = n0; sm.nrmS[s][1] = n1; sm.nrmS[s][2] = n2;

    float ds  = lat[0] + ACT_SHIFT;
    float sp  = fmaxf(ds, 0.f) + log1pf(__expf(-fabsf(ds)));     // softplus
    alpha_s   = inb ? (1.f - __expf(-sp * STEPSZ)) : 0.f;
    sm.alphaS[s] = alpha_s;

    float dvn = -(vx*n0 + vy*n1 + vz*n2);
    float rx = 2.f*dvn*n0 + vx, ry = 2.f*dvn*n1 + vy, rz = 2.f*dvn*n2 + vz;

    float f[64];
#pragma unroll
    for (int c = 0; c < 15; ++c) f[c] = inb ? lat[1 + c] : 0.f;
    f[15] = rx; f[16] = ry; f[17] = rz;
    float fr = 1.f;
#pragma unroll
    for (int q = 0; q < 6; ++q) {
      f[18 + q*3 + 0] = __sinf(rx * fr);
      f[18 + q*3 + 1] = __sinf(ry * fr);
      f[18 + q*3 + 2] = __sinf(rz * fr);
      f[36 + q*3 + 0] = __cosf(rx * fr);
      f[36 + q*3 + 1] = __cosf(ry * fr);
      f[36 + q*3 + 2] = __cosf(rz * fr);
      fr *= 2.f;
    }
#pragma unroll
    for (int j = 54; j < 64; ++j) f[j] = 0.f;

    _Float16* rowp = sm.buf + s * 64;
#pragma unroll
    for (int j = 0; j < 64; j += 2) {
      h2 p; p.x = (_Float16)f[j]; p.y = (_Float16)f[j + 1];
      *(h2*)(rowp + j) = p;
    }
  }
  __syncthreads();

  // ---------------- fused MLP via WMMA: each wave owns 32 rows ----------------
  const int mbase = wv * 32;
  const int arow0 = mbase + (lane & 15);
  const int arow1 = arow0 + 16;
  const int hh    = lane >> 4;

#pragma unroll
  for (int layer = 0; layer < 3; ++layer) {
    // A fragments for both M-tiles (regs)
    v16h a00 = ld_fragA(sm.buf + arow0 * 64, 0, hh);
    v16h a01 = ld_fragA(sm.buf + arow0 * 64, 1, hh);
    v16h a10 = ld_fragA(sm.buf + arow1 * 64, 0, hh);
    v16h a11 = ld_fragA(sm.buf + arow1 * 64, 1, hh);
    // All 8 B fragments of this layer (regs, reused across both M-tiles)
    const int fb = layer * 8;
    v16h bR[8];
#pragma unroll
    for (int q = 0; q < 8; ++q)
      bR[q] = ld_fragB(sm.wfrag + (fb + q) * 512 + lane * 16);
    __syncthreads();                      // operands staged in regs -> safe in-place update
    const float* bs = sm.bias + layer * 64;
#pragma unroll
    for (int mt = 0; mt < 2; ++mt) {
      v16h aA = mt ? a10 : a00;
      v16h aB = mt ? a11 : a01;
#pragma unroll
      for (int nt = 0; nt < 4; ++nt) {
        v8f c = {};
        c = __builtin_amdgcn_wmma_f32_16x16x32_f16(false, aA, false, bR[nt*2+0], (short)0, c, false, false);
        c = __builtin_amdgcn_wmma_f32_16x16x32_f16(false, aB, false, bR[nt*2+1], (short)0, c, false, false);
        int col = nt * 16 + (lane & 15);
        float bv = bs[col];
#pragma unroll
        for (int rr = 0; rr < 8; ++rr) {
          int row = mbase + mt * 16 + rr + 8 * hh;   // C layout: VGPR r -> M = r + 8*(lane/16)
          float v = fmaxf(c[rr] + bv, 0.f);
          sm.buf[row * 64 + col] = (_Float16)v;
        }
      }
    }
    __syncthreads();
  }

  // layer 3: 64 -> 3 (N padded to 16), sigmoid
  {
    v16h a00 = ld_fragA(sm.buf + arow0 * 64, 0, hh);
    v16h a01 = ld_fragA(sm.buf + arow0 * 64, 1, hh);
    v16h a10 = ld_fragA(sm.buf + arow1 * 64, 0, hh);
    v16h a11 = ld_fragA(sm.buf + arow1 * 64, 1, hh);
    v16h bb0 = ld_fragB(sm.wfrag + 24 * 512 + lane * 16);
    v16h bb1 = ld_fragB(sm.wfrag + 25 * 512 + lane * 16);
    const float* bs = sm.bias + 192;
#pragma unroll
    for (int mt = 0; mt < 2; ++mt) {
      v8f c = {};
      c = __builtin_amdgcn_wmma_f32_16x16x32_f16(false, mt ? a10 : a00, false, bb0, (short)0, c, false, false);
      c = __builtin_amdgcn_wmma_f32_16x16x32_f16(false, mt ? a11 : a01, false, bb1, (short)0, c, false, false);
      int col = lane & 15;
      if (col < 3) {
        float bv = bs[col];
#pragma unroll
        for (int rr = 0; rr < 8; ++rr) {
          int row = mbase + mt * 16 + rr + 8 * hh;
          float x = c[rr] + bv;
          sm.rgbS[row][col] = 1.f / (1.f + __expf(-x));
        }
      }
    }
  }
  __syncthreads();

  // ---------------- compositing: log-space scan + reductions ----------------
  {
    const int s = tid;
    float alpha = alpha_s;
    float l = __logf(fmaxf(1.f - alpha, 1e-10f));
    float x = l;
#pragma unroll
    for (int off = 1; off < 32; off <<= 1) {
      float y = __shfl_up(x, (unsigned)off, 32);
      if (lane >= off) x += y;
    }
    if (lane == 31) sm.waveSum[wv] = x;
    __syncthreads();
    float basev = 0.f;
    for (int i = 0; i < wv; ++i) basev += sm.waveSum[i];
    float incl = basev + x;
    float Texc = __expf(incl - l);        // prod_{j<s} clip(1-alpha_j)
    float w    = alpha * Texc;

    out[OFF_W  + (size_t)ray * 256 + s]      = w;
    out[OFF_AIC + (size_t)ray * 257 + s + 1] = __expf(incl);
    if (tid == 0)   out[OFF_AIC + (size_t)ray * 257] = 1.f;
    if (tid == 255) sm.Ttot = __expf(incl);

    float vr = sm.rgbS[s][0], vg = sm.rgbS[s][1], vb = sm.rgbS[s][2];
    float vals[8] = { w*vr, w*vg, w*vb, w*t_s, w, w*nxs, w*nys, w*nzs };
#pragma unroll
    for (int k = 0; k < 8; ++k) {
      float v = vals[k];
#pragma unroll
      for (int off = 16; off > 0; off >>= 1) v += __shfl_down(v, (unsigned)off, 32);
      if (lane == 0) sm.red[wv][k] = v;
    }
    __syncthreads();
    if (tid == 0) {
      float acc[8];
#pragma unroll
      for (int k = 0; k < 8; ++k) {
        float ssum = 0.f;
        for (int i = 0; i < 8; ++i) ssum += sm.red[i][k];
        acc[k] = ssum;
      }
      float Tt = sm.Ttot;
      out[ray * 3 + 0] = acc[0] + Tt * BGV;
      out[ray * 3 + 1] = acc[1] + Tt * BGV;
      out[ray * 3 + 2] = acc[2] + Tt * BGV;
      float dm = acc[3] + Tt * FARV;
      out[OFF_DEPTH + ray] = dm;
      out[OFF_DISP  + ray] = 1.f / dm;
      out[OFF_ACC   + ray] = acc[4];
      out[OFF_NRM + ray * 3 + 0] = acc[5];
      out[OFF_NRM + ray * 3 + 1] = acc[6];
      out[OFF_NRM + ray * 3 + 2] = acc[7];
    }
  }
}

extern "C" void kernel_launch(void* const* d_in, const int* in_sizes, int n_in,
                              void* d_out, int out_size, void* d_ws, size_t ws_size,
                              hipStream_t stream) {
  (void)in_sizes; (void)n_in; (void)out_size; (void)d_ws; (void)ws_size;
  nerf_render_fused<<<N_RAYSC, 256, 0, stream>>>(
      (const float*)d_in[0],  (const float*)d_in[1],  (const float*)d_in[2],
      (const float*)d_in[3],  (const float*)d_in[4],
      (const float*)d_in[5],  (const float*)d_in[6],
      (const float*)d_in[7],  (const float*)d_in[8],
      (const float*)d_in[9],  (const float*)d_in[10],
      (const float*)d_in[11], (const float*)d_in[12],
      (float*)d_out);
}
